// SelfAttention_7404523618607
// MI455X (gfx1250) — matrix operand order repaired and checked by reading, not compiled, not run on hardware
//
#include <hip/hip_runtime.h>

// ---------------------------------------------------------------------------
// Types for CDNA5 WMMA (wave32): v_wmma_f32_16x16x32_bf16
// ---------------------------------------------------------------------------
typedef __attribute__((ext_vector_type(16))) __bf16 bf16x16;
typedef __attribute__((ext_vector_type(8)))  __bf16 bf16x8;
typedef __attribute__((ext_vector_type(4)))  __bf16 bf16x4;
typedef __attribute__((ext_vector_type(8)))  float  f32x8;
typedef __attribute__((ext_vector_type(4)))  unsigned uint32x4;
typedef __attribute__((ext_vector_type(8)))  int      int32x8;
typedef __attribute__((ext_vector_type(4)))  int      int32x4;

#define EMB   1024
#define NHEAD 16
#define HDIM  64
#define SEQ   2048

// GEMM tiling
#define TILE_M 128
#define TILE_N 64
#define KTILE  32
#define ASTR   40   // LDS row stride in bf16 (32 data + 8 pad = TDM pad 4 DW / 16 DW)

// Attention K/V LDS tiles: 64 x 64 bf16, padded row stride 72
#define KVSTR 72

// ---------------------------------------------------------------------------
// LDS byte offset of a generic pointer into __shared__ (flat LDS addr low 32b)
// ---------------------------------------------------------------------------
static __device__ __forceinline__ unsigned lds_u32(const void* p) {
  return (unsigned)(unsigned long long)(uintptr_t)p;
}

// ---------------------------------------------------------------------------
// Async memory->LDS copy (16B per lane), tracked by ASYNCcnt.
// ---------------------------------------------------------------------------
static __device__ __forceinline__ void async_load_b128(unsigned ldsOff,
                                                       const void* g) {
  asm volatile("global_load_async_to_lds_b128 %0, %1, off"
               :
               : "v"(ldsOff), "v"((unsigned long long)(uintptr_t)g)
               : "memory");
}
static __device__ __forceinline__ void wait_async0() {
  asm volatile("s_wait_asynccnt 0x0" ::: "memory");
}

// ---------------------------------------------------------------------------
// TDM: DMA a 2D tile (32 bf16 wide x rows) from global into LDS with row
// padding 32 -> 40 bf16 (pad_interval = 16 DWORDs, pad_amount = 4 DWORDs).
// stride = global row stride in elements.  Issued wave-uniform.
// ---------------------------------------------------------------------------
static __device__ __forceinline__ void tdm_load_tile32(unsigned ldsOff,
                                                       const void* gptr,
                                                       unsigned rows,
                                                       unsigned stride) {
  unsigned long long ga = (unsigned long long)(uintptr_t)gptr;
  uint32x4 g0;
  g0[0] = 1u;                                   // count=1 (valid), no gather
  g0[1] = ldsOff;                               // lds_addr
  g0[2] = (unsigned)ga;                         // global_addr[31:0]
  g0[3] = (unsigned)((ga >> 32) & 0x01FFFFFFu)  // global_addr[56:32]
          | (2u << 30);                         // type=2 (image)
  int32x8 g1;
  g1[0] = (int)((1u << 16)      // data_size = 2 bytes
                | (1u << 20)    // pad_enable
                | (3u << 22)    // pad_interval: 16 DWORDs
                | (3u << 25));  // pad_amount: 4 DWORDs
  g1[1] = (int)((stride & 0xFFFFu) << 16);          // tensor_dim0[15:0]
  g1[2] = (int)(((stride >> 16) & 0xFFFFu)          // tensor_dim0[31:16]
                | (0xFFFFu << 16));                 // tensor_dim1[15:0]=65535
  g1[3] = (int)(32u << 16);                         // tile_dim0 = 32
  g1[4] = (int)rows;                                // tile_dim1 = rows
  g1[5] = (int)stride;                              // tensor_dim0_stride[31:0]
  g1[6] = 0;
  g1[7] = 0;
  int32x4 z4 = {0, 0, 0, 0};
#if __clang_major__ >= 23
  int32x8 z8 = {0, 0, 0, 0, 0, 0, 0, 0};
  __builtin_amdgcn_tensor_load_to_lds(g0, g1, z4, z4, z8, 0);
#else
  __builtin_amdgcn_tensor_load_to_lds(g0, g1, z4, z4, 0);
#endif
}

// A-fragment (16x32 bf16, ISA layout): lane L -> row L%16,
// K = {kb..kb+7} U {kb+16..kb+23}, kb = (L>=16)*8.
__device__ __forceinline__ bf16x16 frag_a(const __bf16* rowPtr, int kb) {
  bf16x8 lo = *(const bf16x8*)(rowPtr + kb);
  bf16x8 hi = *(const bf16x8*)(rowPtr + kb + 16);
  bf16x16 r;
#pragma unroll
  for (int i = 0; i < 8; ++i) { r[i] = lo[i]; r[i + 8] = hi[i]; }
  return r;
}

// B-fragment (32x16 bf16, ISA layout): lane L -> column L%16,
// K = {ks..ks+7} U {ks+16..ks+23}, ks = (L>=16)*8.
__device__ __forceinline__ bf16x16 frag_b(const __bf16* rowPtr, int ks) {
  bf16x8 lo = *(const bf16x8*)(rowPtr + ks);
  bf16x8 hi = *(const bf16x8*)(rowPtr + ks + 16);
  bf16x16 r;
#pragma unroll
  for (int i = 0; i < 8; ++i) { r[i] = lo[i]; r[i + 8] = hi[i]; }
  return r;
}

__device__ __forceinline__ f32x8 wmma_bf16f32(bf16x16 a, bf16x16 b, f32x8 c) {
  return __builtin_amdgcn_wmma_f32_16x16x32_bf16(false, a, false, b, (short)0, c,
                                                 false, false);
}

// ---------------------------------------------------------------------------
// Weight conversion: W[K][N] fp32 -> Bt[N][K] bf16 (pre-transposed)
// ---------------------------------------------------------------------------
__global__ void cvt_transpose_bf16(const float* __restrict__ W,
                                   __bf16* __restrict__ Bt, int K, int N) {
  int idx = blockIdx.x * blockDim.x + threadIdx.x;
  if (idx >= K * N) return;
  int k = idx / N;
  int n = idx - k * N;
  Bt[(size_t)n * K + k] = (__bf16)W[idx];
}

// ---------------------------------------------------------------------------
// WMMA GEMM: C[M][N] = A[M][K] * Bt[N][K]^T + bias[N]
//   A_F32: A fp32, converted to bf16 by VALU staging; else A bf16 via TDM.
//   B tile always staged by TDM (wave 0).
//   MODE 0: scatter into Q[b,h,s,d], K[b,h,s,d], V^T[b,h,d,s] (bf16)
//   MODE 1: write fp32 out[M][N]
// Block: 256 threads = 8 waves (4 x 2), block tile 128x64, wave tile 32x32.
// ---------------------------------------------------------------------------
template <bool A_F32, int MODE>
__global__ __launch_bounds__(256) void gemm_bf16_wmma(
    const void* __restrict__ Ap, const __bf16* __restrict__ Bt,
    const float* __restrict__ bias, float* __restrict__ outF,
    __bf16* __restrict__ Qo, __bf16* __restrict__ Ko, __bf16* __restrict__ Vo,
    int M, int N, int K) {
  __shared__ __bf16 ldsA[TILE_M * ASTR];
  __shared__ __bf16 ldsB[TILE_N * ASTR];

  const int tid  = threadIdx.x;
  const int lane = tid & 31;
  const int wid  = tid >> 5;
  const int wm   = wid & 3;   // 0..3  -> 32-row strip
  const int wn   = wid >> 2;  // 0..1  -> 32-col strip
  const int m0   = blockIdx.x * TILE_M;
  const int n0   = blockIdx.y * TILE_N;

  const int nl  = lane & 15;
  const int kbA = (lane >> 4) * 8;   // A-fragment K base
  const int ksB = (lane >> 4) * 8;   // B-fragment K base

  f32x8 acc[2][2];
#pragma unroll
  for (int i = 0; i < 2; ++i)
#pragma unroll
    for (int j = 0; j < 2; ++j)
#pragma unroll
      for (int r = 0; r < 8; ++r) acc[i][j][r] = 0.0f;

  const float*  Af = (const float*)Ap;
  const __bf16* Ab = (const __bf16*)Ap;

  const int ra = tid >> 3;        // 0..31
  const int ca = (tid & 7) * 4;   // 0,4,...,28 (K within tile)

  for (int kt = 0; kt < K; kt += KTILE) {
    // --- stage B tile (64 x 32) via Tensor Data Mover ---
    if (wid == 0) {
      tdm_load_tile32(lds_u32(ldsB), Bt + (size_t)n0 * K + kt, TILE_N,
                      (unsigned)K);
      __builtin_amdgcn_s_wait_tensorcnt(0);
    }
    if constexpr (A_F32) {
      // --- stage A tile (128 x 32) with fp32 -> bf16 conversion ---
#pragma unroll
      for (int i = 0; i < 4; ++i) {
        int r = ra + 32 * i;
        const float4 f = *(const float4*)(Af + (size_t)(m0 + r) * K + kt + ca);
        bf16x4 v;
        v[0] = (__bf16)f.x; v[1] = (__bf16)f.y;
        v[2] = (__bf16)f.z; v[3] = (__bf16)f.w;
        *(bf16x4*)(&ldsA[r * ASTR + ca]) = v;
      }
    } else {
      // --- stage A tile (128 x 32) via TDM (already bf16) ---
      if (wid == 1) {
        tdm_load_tile32(lds_u32(ldsA), Ab + (size_t)m0 * K + kt, TILE_M,
                        (unsigned)K);
        __builtin_amdgcn_s_wait_tensorcnt(0);
      }
    }
    __syncthreads();

    bf16x16 af[2], bfr[2];
#pragma unroll
    for (int t = 0; t < 2; ++t) {
      af[t]  = frag_a(&ldsA[(wm * 32 + t * 16 + nl) * ASTR], kbA);
      bfr[t] = frag_b(&ldsB[(wn * 32 + t * 16 + nl) * ASTR], ksB);
    }
#pragma unroll
    for (int tm = 0; tm < 2; ++tm)
#pragma unroll
      for (int tn = 0; tn < 2; ++tn)
        acc[tm][tn] = wmma_bf16f32(af[tm], bfr[tn], acc[tm][tn]);
    __syncthreads();
  }

  // --- epilogue (C layout: VGPR r, lanes 0-15 -> M=r, lanes 16-31 -> M=r+8) ---
  const int hh = lane >> 4;
#pragma unroll
  for (int tm = 0; tm < 2; ++tm) {
#pragma unroll
    for (int tn = 0; tn < 2; ++tn) {
#pragma unroll
      for (int r = 0; r < 8; ++r) {
        int row = m0 + wm * 32 + tm * 16 + r + hh * 8;
        int col = n0 + wn * 32 + tn * 16 + nl;
        float v = acc[tm][tn][r] + bias[col];
        if (MODE == 0) {
          int bb = row >> 11, s = row & (SEQ - 1);
          int j = col;
          if (j < EMB) {
            int h = j >> 6, d = j & 63;
            Qo[(((size_t)(bb * NHEAD + h)) * SEQ + s) * HDIM + d] = (__bf16)v;
          } else if (j < 2 * EMB) {
            j -= EMB;
            int h = j >> 6, d = j & 63;
            Ko[(((size_t)(bb * NHEAD + h)) * SEQ + s) * HDIM + d] = (__bf16)v;
          } else {
            j -= 2 * EMB;
            int h = j >> 6, d = j & 63;
            Vo[(((size_t)(bb * NHEAD + h)) * HDIM + d) * SEQ + s] = (__bf16)v;
          }
        } else {
          outF[(size_t)row * N + col] = v;
        }
      }
    }
  }
}

// ---------------------------------------------------------------------------
// Flash-style causal attention, all matmuls via WMMA.
// Block = 128 threads = 4 waves; each wave owns a 16-query strip of a
// 64-query block. K and V^T tiles are staged cooperatively into LDS ONCE per
// key block using async memory->LDS copies (removes the 4x per-wave duplicate
// global traffic), then fragments are read from LDS.
//   Q,K: [b,h,s,d] bf16;  V^T: [b,h,d,s] bf16;  Y: [b*s][h*64+d] bf16.
// ---------------------------------------------------------------------------
__global__ __launch_bounds__(128) void attn_wmma(const __bf16* __restrict__ Qb,
                                                 const __bf16* __restrict__ Kb,
                                                 const __bf16* __restrict__ Vt,
                                                 __bf16* __restrict__ Y) {
  __shared__ __bf16 ptile[4 * 16 * KVSTR];  // per-wave 16x64 P tile
  __shared__ __bf16 ldsK[64 * KVSTR];       // 64 keys x 64 d
  __shared__ __bf16 ldsV[64 * KVSTR];       // 64 d   x 64 s

  const int tid  = threadIdx.x;
  const int lane = tid & 31;
  const int wid  = tid >> 5;
  const int qb   = blockIdx.x;  // 0..31
  const int h    = blockIdx.y;  // 0..15
  const int b    = blockIdx.z;  // 0..3

  const size_t  bh = (size_t)(b * NHEAD + h);
  const __bf16* Qh = Qb + bh * SEQ * HDIM;
  const __bf16* Kh = Kb + bh * SEQ * HDIM;
  const __bf16* Vh = Vt + bh * HDIM * SEQ;

  const int q0  = qb * 64 + wid * 16;
  const int nl  = lane & 15;
  const int hh  = lane >> 4;
  const int kbA = hh * 8;
  const int ksB = hh * 8;

  // Q fragments for this wave's 16 rows (d chunks 0..31 and 32..63)
  bf16x16 qf[2];
  {
    const __bf16* rp = Qh + (size_t)(q0 + nl) * HDIM;
    qf[0] = frag_a(rp, kbA);
    qf[1] = frag_a(rp + 32, kbA);
  }

  f32x8 yacc[4];
#pragma unroll
  for (int dt = 0; dt < 4; ++dt)
#pragma unroll
    for (int r = 0; r < 8; ++r) yacc[dt][r] = 0.0f;

  float mrun[8], lrun[8];
#pragma unroll
  for (int r = 0; r < 8; ++r) { mrun[r] = -1.0e30f; lrun[r] = 0.0f; }

  __bf16* myP = &ptile[wid * 16 * KVSTR];

  for (int kblk = 0; kblk <= qb; ++kblk) {
    const int kbase = kblk * 64;
    const bool diag = (kblk == qb);

    // ---- async-stage K (64x64) and V^T (64x64) tiles into LDS ----
    // 512 16B chunks per tile; 128 threads x 4 chunks each.
#pragma unroll
    for (int i = 0; i < 4; ++i) {
      const int c   = tid + 128 * i;     // 0..511
      const int row = c >> 3;            // 0..63
      const int c16 = (c & 7) * 8;       // bf16 offset of 16B chunk
      async_load_b128(lds_u32(&ldsK[row * KVSTR + c16]),
                      Kh + (size_t)(kbase + row) * HDIM + c16);
      async_load_b128(lds_u32(&ldsV[row * KVSTR + c16]),
                      Vh + (size_t)row * SEQ + kbase + c16);
    }
    wait_async0();
    __syncthreads();

    // ---- scores: S = Q K^T (16 x 64 over this key block) ----
    f32x8 sc[4];
#pragma unroll
    for (int nt = 0; nt < 4; ++nt) {
#pragma unroll
      for (int r = 0; r < 8; ++r) sc[nt][r] = 0.0f;
      const __bf16* krp = &ldsK[(nt * 16 + nl) * KVSTR];
      bf16x16 kf = frag_b(krp, ksB);
      sc[nt] = wmma_bf16f32(qf[0], kf, sc[nt]);
      kf = frag_b(krp + 32, ksB);
      sc[nt] = wmma_bf16f32(qf[1], kf, sc[nt]);
    }

    // ---- scale 1/sqrt(64), causal mask, row max ----
    float rmax[8];
#pragma unroll
    for (int r = 0; r < 8; ++r) rmax[r] = -1.0e30f;
#pragma unroll
    for (int nt = 0; nt < 4; ++nt) {
      const int key = kbase + nt * 16 + nl;
#pragma unroll
      for (int r = 0; r < 8; ++r) {
        const int qrow = q0 + r + hh * 8;
        float s = sc[nt][r] * 0.125f;
        if (diag && key > qrow) s = -1.0e30f;
        sc[nt][r] = s;
        rmax[r] = fmaxf(rmax[r], s);
      }
    }
#pragma unroll
    for (int r = 0; r < 8; ++r) {
      float v = rmax[r];
      v = fmaxf(v, __shfl_xor(v, 1, 32));
      v = fmaxf(v, __shfl_xor(v, 2, 32));
      v = fmaxf(v, __shfl_xor(v, 4, 32));
      v = fmaxf(v, __shfl_xor(v, 8, 32));
      rmax[r] = v;
    }

    float alpha[8], rsum[8];
#pragma unroll
    for (int r = 0; r < 8; ++r) {
      float mnew = fmaxf(mrun[r], rmax[r]);
      alpha[r] = __expf(mrun[r] - mnew);
      mrun[r] = mnew;
      rsum[r] = 0.0f;
    }

    // ---- P = exp(S - m), stash bf16 P into wave-private LDS ----
#pragma unroll
    for (int nt = 0; nt < 4; ++nt) {
#pragma unroll
      for (int r = 0; r < 8; ++r) {
        float p = __expf(sc[nt][r] - mrun[r]);
        rsum[r] += p;
        myP[(r + hh * 8) * KVSTR + nt * 16 + nl] = (__bf16)p;
      }
    }
#pragma unroll
    for (int r = 0; r < 8; ++r) {
      float v = rsum[r];
      v += __shfl_xor(v, 1, 32);
      v += __shfl_xor(v, 2, 32);
      v += __shfl_xor(v, 4, 32);
      v += __shfl_xor(v, 8, 32);
      lrun[r] = lrun[r] * alpha[r] + v;
    }
#pragma unroll
    for (int dt = 0; dt < 4; ++dt)
#pragma unroll
      for (int r = 0; r < 8; ++r) yacc[dt][r] *= alpha[r];

    // ---- Y += P V  (P as A-fragments, V from LDS, contiguous in s) ----
    bf16x16 pf0 = frag_a(&myP[nl * KVSTR], kbA);
    bf16x16 pf1 = frag_a(&myP[nl * KVSTR + 32], kbA);
#pragma unroll
    for (int dt = 0; dt < 4; ++dt) {
      const __bf16* vrp = &ldsV[(dt * 16 + nl) * KVSTR];
      bf16x16 vf = frag_b(vrp, ksB);
      yacc[dt] = wmma_bf16f32(pf0, vf, yacc[dt]);
      vf = frag_b(vrp + 32, ksB);
      yacc[dt] = wmma_bf16f32(pf1, vf, yacc[dt]);
    }
    __syncthreads();  // protect ldsK/ldsV before next block's staging
  }

  // ---- normalize and write Y[b*s][h*64+d] as bf16 ----
#pragma unroll
  for (int r = 0; r < 8; ++r) {
    const float inv = 1.0f / lrun[r];
    const int token = b * SEQ + q0 + r + hh * 8;
#pragma unroll
    for (int dt = 0; dt < 4; ++dt) {
      Y[(size_t)token * EMB + h * HDIM + dt * 16 + nl] =
          (__bf16)(yacc[dt][r] * inv);
    }
  }
}

// ---------------------------------------------------------------------------
// Launch: cvt weights -> GEMM(QKV, scatter) -> attention -> GEMM(proj, fp32)
// Workspace layout (72 MB total):
//   [0,16M)  ybf      (attention output, bf16)
//   [16,22M) WattnT   [3072][1024] bf16
//   [22,24M) WprojT   [1024][1024] bf16
//   [24,40M) Q  [b,h,s,d] bf16
//   [40,56M) K  [b,h,s,d] bf16
//   [56,72M) V^T[b,h,d,s] bf16
// ---------------------------------------------------------------------------
extern "C" void kernel_launch(void* const* d_in, const int* in_sizes, int n_in,
                              void* d_out, int out_size, void* d_ws,
                              size_t ws_size, hipStream_t stream) {
  const float* x  = (const float*)d_in[0];
  const float* Wa = (const float*)d_in[1];
  const float* ba = (const float*)d_in[2];
  const float* Wp = (const float*)d_in[3];
  const float* bp = (const float*)d_in[4];
  float* out = (float*)d_out;

  char* ws = (char*)d_ws;
  const size_t MB = 1024 * 1024;
  __bf16* ybf = (__bf16*)(ws + 0);
  __bf16* WaT = (__bf16*)(ws + 16 * MB);
  __bf16* WpT = (__bf16*)(ws + 22 * MB);
  __bf16* Qb  = (__bf16*)(ws + 24 * MB);
  __bf16* Kb  = (__bf16*)(ws + 40 * MB);
  __bf16* Vt  = (__bf16*)(ws + 56 * MB);

  cvt_transpose_bf16<<<(EMB * 3 * EMB) / 256, 256, 0, stream>>>(Wa, WaT, EMB,
                                                                3 * EMB);
  cvt_transpose_bf16<<<(EMB * EMB) / 256, 256, 0, stream>>>(Wp, WpT, EMB, EMB);

  // QKV: [8192 x 3072] = x[8192 x 1024] * W_attn, scattered into Q/K/V^T
  gemm_bf16_wmma<true, 0><<<dim3(8192 / TILE_M, 3072 / TILE_N), 256, 0,
                            stream>>>(x, WaT, ba, nullptr, Qb, Kb, Vt, 8192,
                                      3072, EMB);

  // causal flash attention per (q-block, head, batch)
  attn_wmma<<<dim3(SEQ / 64, NHEAD, 4), 128, 0, stream>>>(Qb, Kb, Vt, ybf);

  // out: [8192 x 1024] fp32 = ybf * W_proj + b_proj
  gemm_bf16_wmma<false, 1><<<dim3(8192 / TILE_M, 1024 / TILE_N), 256, 0,
                             stream>>>(ybf, WpT, bp, out, nullptr, nullptr,
                                       nullptr, 8192, EMB, EMB);
}
